// PatchedCausalSelfAttention_64750926954930
// MI455X (gfx1250) — matrix operand order repaired and checked by reading, not compiled, not run on hardware
//
#include <hip/hip_runtime.h>
#include <hip/hip_bf16.h>
#include <math.h>

// ---------------------------------------------------------------------------
// PatchedCausalSelfAttention for MI455X (gfx1250, wave32, WMMA f16->f32)
// B=4 S=2048 D=768 DC=384 H=12 HD=32
//
//   WqkvT  [1152][768] f16 = (W_attn_{q,k,v} @ P_{q,k,v})^T       (fold kernel)
//   WoutT  [768][384]  f16 = (W_back @ W_proj)^T                  (fold kernel)
//   Q,K    [BH][S][32] f16, Vt [BH][32][S] f16                    (qkv GEMM, WMMA 2Mx3N)
//   Ctx    [B*S][384]  f16  via per-head flash attention          (WMMA + async->LDS)
//   out    [B*S][768]  f32 = Ctx @ W_out + b_proj                 (WMMA 2Mx3N)
// ---------------------------------------------------------------------------

typedef _Float16 half_t;
typedef __attribute__((ext_vector_type(16))) _Float16 v16h;
typedef __attribute__((ext_vector_type(8)))  float    v8f;

#define WMMA_F16(a, b, c) \
  __builtin_amdgcn_wmma_f32_16x16x32_f16(false, (a), false, (b), (short)0, (c), false, false)

#define NB   4
#define NS   2048
#define ND   768
#define NDC  384
#define NH   12
#define NHD  32
#define NROWS (NB * NS)          // 8192
#define N3D  (3 * NDC)           // 1152

// CDNA5 async global->LDS path (guarded: falls back to direct global loads)
#if defined(__AMDGCN__) && \
    __has_builtin(__builtin_amdgcn_global_load_async_to_lds_b128) && \
    __has_builtin(__builtin_amdgcn_s_wait_asynccnt)
#define ASYNC_LDS 1
// probe-derived signature: (v4i addrspace(1)*, v4i addrspace(3)*, imm offset, imm cpol)
typedef int v4i_vs __attribute__((vector_size(16)));
typedef __attribute__((address_space(1))) v4i_vs* gvp_t;
typedef __attribute__((address_space(3))) v4i_vs* lvp_t;
__device__ __forceinline__ void async_cp16(const void* g, void* l) {
  __builtin_amdgcn_global_load_async_to_lds_b128((gvp_t)g, (lvp_t)l, 0, 0);
}
#endif

// ---- fragment loaders (layouts per CDNA5 ISA §7.12.2) ----------

// A 16x32 f16: lane L<16 -> row L, elems 0..7=K0..7, 8..15=K16..23
//              lane L>=16 -> row L-16, elems 0..7=K8..15, 8..15=K24..31
__device__ __forceinline__ v16h load_a_frag(const half_t* __restrict__ base,
                                            int row_stride) {
  const int lane = threadIdx.x & 31;
  const half_t* p = base + (size_t)(lane & 15) * row_stride + ((lane >> 4) << 3);
  v16h a;
#pragma unroll
  for (int i = 0; i < 8; ++i) a[i] = p[i];
#pragma unroll
  for (int i = 0; i < 8; ++i) a[8 + i] = p[16 + i];
  return a;
}

// same, converting fp32 on the fly (v_cvt_pk_f16_f32)
__device__ __forceinline__ v16h load_a_frag_f32(const float* __restrict__ rowptr) {
  const int lane = threadIdx.x & 31;
  const float* p = rowptr + ((lane >> 4) << 3);
  v16h a;
#pragma unroll
  for (int i = 0; i < 8; ++i) a[i] = (half_t)p[i];
#pragma unroll
  for (int i = 0; i < 8; ++i) a[8 + i] = (half_t)p[16 + i];
  return a;
}

// B 32x16 f16: lane L -> col N=L%16; elems 0..7=K(L>=16?8:0)+i, 8..15=K16+(L>=16?8:0)+i.
// Each column's K values contiguous (stride col_stride between columns).
__device__ __forceinline__ v16h load_b_frag(const half_t* __restrict__ base,
                                            int col_stride) {
  const int lane = threadIdx.x & 31;
  const half_t* p = base + (size_t)(lane & 15) * col_stride + ((lane >> 4) << 3);
  v16h b;
#pragma unroll
  for (int i = 0; i < 8; ++i) b[i] = p[i];
#pragma unroll
  for (int i = 0; i < 8; ++i) b[8 + i] = p[16 + i];
  return b;
}

// ---- kernel 1: fold weights (tiny, scalar fp32, writes f16) ---------------
__global__ __launch_bounds__(256) void fold_kernel(
    const float* __restrict__ W_attn, const float* __restrict__ b_attn,
    const float* __restrict__ P_q, const float* __restrict__ P_k,
    const float* __restrict__ P_v, const float* __restrict__ W_back,
    const float* __restrict__ W_proj,
    half_t* __restrict__ WqkvT, half_t* __restrict__ WoutT,
    float* __restrict__ b_eff) {
  const int NQKV = N3D * ND;
  const int NOUT = ND * NDC;
  int idx = blockIdx.x * blockDim.x + threadIdx.x;
  if (idx < NQKV) {
    int n = idx / ND, d = idx % ND;
    int sel = n / NDC, j = n % NDC;
    const float* P = (sel == 0) ? P_q : ((sel == 1) ? P_k : P_v);
    const float* wrow = W_attn + (size_t)d * 2304 + sel * ND;
    float acc = 0.f;
    for (int t = 0; t < ND; ++t) acc += wrow[t] * P[(size_t)t * NDC + j];
    WqkvT[(size_t)n * ND + d] = (half_t)acc;          // [n][d]
  } else if (idx < NQKV + NOUT) {
    int e = idx - NQKV;
    int n = e / NDC, c = e % NDC;                     // W_out[c][n] -> [n][c]
    float acc = 0.f;
    for (int t = 0; t < ND; ++t)
      acc += W_back[(size_t)c * ND + t] * W_proj[(size_t)t * ND + n];
    WoutT[(size_t)n * NDC + c] = (half_t)acc;
  } else if (idx < NQKV + NOUT + N3D) {
    int n = idx - NQKV - NOUT;
    int sel = n / NDC, j = n % NDC;
    const float* P = (sel == 0) ? P_q : ((sel == 1) ? P_k : P_v);
    float acc = 0.f;
    for (int t = 0; t < ND; ++t) acc += b_attn[sel * ND + t] * P[(size_t)t * NDC + j];
    b_eff[n] = acc;
  }
}

// ---- kernel 2: qkv GEMM, 2Mx3N register-blocked per wave ------------------
// grid (8192/32, 72/12) x 128 thr: block = 32 rows x 12 col-tiles
__global__ __launch_bounds__(128) void qkv_gemm_kernel(
    const float* __restrict__ X, const half_t* __restrict__ WqkvT,
    const float* __restrict__ b_eff,
    half_t* __restrict__ Qh, half_t* __restrict__ Kh, half_t* __restrict__ Vt) {
  const int wave = threadIdx.x >> 5;
  const int lane = threadIdx.x & 31;
  const int nt0  = (blockIdx.y * 4 + wave) * 3;       // first of 3 N tiles
  const float* xr0 = X + (size_t)(blockIdx.x * 32 + (lane & 15)) * ND;
  const float* xr1 = xr0 + (size_t)16 * ND;
  const half_t* w0 = WqkvT + (size_t)(nt0 + 0) * 16 * ND;
  const half_t* w1 = WqkvT + (size_t)(nt0 + 1) * 16 * ND;
  const half_t* w2 = WqkvT + (size_t)(nt0 + 2) * 16 * ND;

  v8f a00 = {}, a01 = {}, a02 = {}, a10 = {}, a11 = {}, a12 = {};
#pragma unroll 2
  for (int k0 = 0; k0 < ND; k0 += 32) {
    __builtin_prefetch(w0 + k0 + 64, 0, 1);
    v16h fa0 = load_a_frag_f32(xr0 + k0);
    v16h fa1 = load_a_frag_f32(xr1 + k0);
    v16h fb0 = load_b_frag(w0 + k0, ND);
    v16h fb1 = load_b_frag(w1 + k0, ND);
    v16h fb2 = load_b_frag(w2 + k0, ND);
    a00 = WMMA_F16(fa0, fb0, a00);
    a10 = WMMA_F16(fa1, fb0, a10);
    a01 = WMMA_F16(fa0, fb1, a01);
    a11 = WMMA_F16(fa1, fb1, a11);
    a02 = WMMA_F16(fa0, fb2, a02);
    a12 = WMMA_F16(fa1, fb2, a12);
  }

  const int Mbase = (lane >> 4) << 3;
  auto store_tile = [&](v8f acc, int mi, int j) {
    const int n = (nt0 + j) * 16 + (lane & 15);
    const float bias = b_eff[n];
#pragma unroll
    for (int r = 0; r < 8; ++r) {
      int row_g = blockIdx.x * 32 + mi * 16 + Mbase + r;
      int bi = row_g >> 11, s = row_g & (NS - 1);
      half_t hv = (half_t)(acc[r] + bias);
      if (n < NDC) {                                  // Q: [BH][S][32]
        int h = n >> 5, hd = n & 31;
        Qh[(((size_t)(bi * NH + h) * NS + s) << 5) + hd] = hv;
      } else if (n < 2 * NDC) {                       // K: [BH][S][32]
        int nn = n - NDC; int h = nn >> 5, hd = nn & 31;
        Kh[(((size_t)(bi * NH + h) * NS + s) << 5) + hd] = hv;
      } else {                                        // V^T: [BH][32][S]
        int nn = n - 2 * NDC; int h = nn >> 5, hd = nn & 31;
        Vt[((size_t)(bi * NH + h) * NHD + hd) * NS + s] = hv;
      }
    }
  };
  store_tile(a00, 0, 0); store_tile(a10, 1, 0);
  store_tile(a01, 0, 1); store_tile(a11, 1, 1);
  store_tile(a02, 0, 2); store_tile(a12, 1, 2);
}

// ---- kernel 3: causal flash attention, async K/V double-buffer in LDS -----
__global__ __launch_bounds__(32) void attn_kernel(
    const half_t* __restrict__ Qh, const half_t* __restrict__ Kh,
    const half_t* __restrict__ Vt, half_t* __restrict__ Ctx) {
  __shared__ alignas(16) half_t Ktile[2][32 * 32];    // [buf][key][hd]
  __shared__ alignas(16) half_t Vtile[2][32 * 32];    // [buf][hd][key]
  __shared__ alignas(16) half_t Pbuf[16 * 32];        // probs C-layout -> A-layout
  const int bh    = blockIdx.x;                       // 0..47
  const int qb    = blockIdx.y * 16;
  const int lane  = threadIdx.x & 31;
  const int Mbase = (lane >> 4) << 3;

  const half_t* Kbase = Kh + (size_t)bh * NS * NHD;
  const half_t* Vbase = Vt + (size_t)bh * NHD * NS;
  const v16h qa = load_a_frag(Qh + (size_t)bh * NS * NHD + (size_t)qb * NHD, NHD);
  const float scale = 0.17677669529663687f;           // 1/sqrt(32)

#ifdef ASYNC_LDS
  auto stage_kv = [&](int kb, int buf) {
    // K tile: 2048 contiguous bytes -> 4 async b128 ops (32 lanes x 16B)
    const char* kg = (const char*)(Kbase + (size_t)kb * NHD) + lane * 16;
    char* kl = (char*)Ktile[buf] + lane * 16;
    async_cp16(kg, kl);
    async_cp16(kg + 512, kl + 512);
    async_cp16(kg + 1024, kl + 1024);
    async_cp16(kg + 1536, kl + 1536);
    // V tile: 32 rows of 64B (stride S*2B) -> 4 async b128 ops, 8 rows each
    const int hd = lane >> 2, ch = lane & 3;
    const char* vg = (const char*)(Vbase + (size_t)hd * NS + kb) + ch * 16;
    char* vl = (char*)Vtile[buf] + hd * 64 + ch * 16;
    const size_t gstep = (size_t)8 * NS * sizeof(half_t);
    async_cp16(vg, vl);
    async_cp16(vg + gstep, vl + 512);
    async_cp16(vg + 2 * gstep, vl + 1024);
    async_cp16(vg + 3 * gstep, vl + 1536);
  };
#endif

  float m[8], l[8];
#pragma unroll
  for (int r = 0; r < 8; ++r) { m[r] = -3.0e38f; l[r] = 0.f; }
  v8f o0 = {}, o1 = {};

  const int nkb = (qb + 15) / 32 + 1;                 // 32-key tiles (uniform)
#ifdef ASYNC_LDS
  stage_kv(0, 0);
#endif
  for (int t = 0; t < nkb; ++t) {
    const int kb = t * 32;
#ifdef ASYNC_LDS
    if (t + 1 < nkb) {
      stage_kv(kb + 32, (t + 1) & 1);
      __builtin_amdgcn_s_wait_asynccnt(8);            // tile t done, t+1 in flight
    } else {
      __builtin_amdgcn_s_wait_asynccnt(0);
    }
    const half_t* Kt = Ktile[t & 1];
    const half_t* Vl = Vtile[t & 1];
    v16h kf0 = load_b_frag(Kt, 32);                   // keys kb..kb+15
    v16h kf1 = load_b_frag(Kt + 16 * 32, 32);         // keys kb+16..kb+31
#else
    v16h kf0 = load_b_frag(Kbase + (size_t)kb * NHD, NHD);
    v16h kf1 = load_b_frag(Kbase + (size_t)(kb + 16) * NHD, NHD);
#endif
    v8f z = {};
    v8f s0 = WMMA_F16(qa, kf0, z);
    v8f s1 = WMMA_F16(qa, kf1, z);

    const int kg0 = kb + (lane & 15);
    const int kg1 = kg0 + 16;
#pragma unroll
    for (int r = 0; r < 8; ++r) {
      const int qg = qb + Mbase + r;
      float p0 = (kg0 <= qg) ? s0[r] * scale : -10000.0f;  // ref masks to -1e4
      float p1 = (kg1 <= qg) ? s1[r] * scale : -10000.0f;
      float rm = fmaxf(p0, p1);                      // row reduce in 16-lane group
      rm = fmaxf(rm, __shfl_xor(rm, 1, 32));
      rm = fmaxf(rm, __shfl_xor(rm, 2, 32));
      rm = fmaxf(rm, __shfl_xor(rm, 4, 32));
      rm = fmaxf(rm, __shfl_xor(rm, 8, 32));
      float mn = fmaxf(m[r], rm);
      float alpha = __expf(m[r] - mn);
      float e0 = __expf(p0 - mn);
      float e1 = __expf(p1 - mn);
      float rs = e0 + e1;
      rs += __shfl_xor(rs, 1, 32);
      rs += __shfl_xor(rs, 2, 32);
      rs += __shfl_xor(rs, 4, 32);
      rs += __shfl_xor(rs, 8, 32);
      l[r] = l[r] * alpha + rs;
      m[r] = mn;
      o0[r] *= alpha;
      o1[r] *= alpha;
      const int M = Mbase + r;
      Pbuf[M * 32 + (lane & 15)]      = (half_t)e0;
      Pbuf[M * 32 + 16 + (lane & 15)] = (half_t)e1;
    }
    __syncthreads();
    v16h pa = load_a_frag(Pbuf, 32);                  // P as A-frag (K=32 keys)
#ifdef ASYNC_LDS
    v16h vf0 = load_b_frag(Vl, 32);                   // hd 0..15
    v16h vf1 = load_b_frag(Vl + 16 * 32, 32);         // hd 16..31
#else
    v16h vf0 = load_b_frag(Vbase + kb, NS);
    v16h vf1 = load_b_frag(Vbase + (size_t)16 * NS + kb, NS);
#endif
    o0 = WMMA_F16(pa, vf0, o0);
    o1 = WMMA_F16(pa, vf1, o1);
    __syncthreads();
  }

  const int bi = bh / NH, h = bh % NH;
#pragma unroll
  for (int r = 0; r < 8; ++r) {
    float inv = 1.0f / l[r];
    size_t row_g = (size_t)bi * NS + qb + Mbase + r;
    int c0 = h * NHD + (lane & 15);
    Ctx[row_g * NDC + c0]      = (half_t)(o0[r] * inv);
    Ctx[row_g * NDC + c0 + 16] = (half_t)(o1[r] * inv);
  }
}

// ---- kernel 4: output GEMM, 2Mx3N register-blocked ------------------------
// grid (8192/32, 48/12) x 128 thr
__global__ __launch_bounds__(128) void out_gemm_kernel(
    const half_t* __restrict__ Ctx, const half_t* __restrict__ WoutT,
    const float* __restrict__ b_proj, float* __restrict__ Y) {
  const int wave = threadIdx.x >> 5;
  const int lane = threadIdx.x & 31;
  const int nt0  = (blockIdx.y * 4 + wave) * 3;
  const half_t* ar0 = Ctx + (size_t)(blockIdx.x * 32 + (lane & 15)) * NDC;
  const half_t* ar1 = ar0 + (size_t)16 * NDC;
  const half_t* w0 = WoutT + (size_t)(nt0 + 0) * 16 * NDC;
  const half_t* w1 = WoutT + (size_t)(nt0 + 1) * 16 * NDC;
  const half_t* w2 = WoutT + (size_t)(nt0 + 2) * 16 * NDC;

  v8f a00 = {}, a01 = {}, a02 = {}, a10 = {}, a11 = {}, a12 = {};
#pragma unroll 2
  for (int k0 = 0; k0 < NDC; k0 += 32) {
    // lane row folded into ar0/ar1 -> stride 0 keeps only the K-offset logic
    v16h fa0 = load_a_frag(ar0 + k0, 0);
    v16h fa1 = load_a_frag(ar1 + k0, 0);
    v16h fb0 = load_b_frag(w0 + k0, NDC);
    v16h fb1 = load_b_frag(w1 + k0, NDC);
    v16h fb2 = load_b_frag(w2 + k0, NDC);
    a00 = WMMA_F16(fa0, fb0, a00);
    a10 = WMMA_F16(fa1, fb0, a10);
    a01 = WMMA_F16(fa0, fb1, a01);
    a11 = WMMA_F16(fa1, fb1, a11);
    a02 = WMMA_F16(fa0, fb2, a02);
    a12 = WMMA_F16(fa1, fb2, a12);
  }

  const int Mbase = (lane >> 4) << 3;
  auto store_tile = [&](v8f acc, int mi, int j) {
    const int n = (nt0 + j) * 16 + (lane & 15);
    const float bias = b_proj[n];
#pragma unroll
    for (int r = 0; r < 8; ++r) {
      size_t row_g = (size_t)blockIdx.x * 32 + mi * 16 + Mbase + r;
      Y[row_g * ND + n] = acc[r] + bias;
    }
  };
  store_tile(a00, 0, 0); store_tile(a10, 1, 0);
  store_tile(a01, 0, 1); store_tile(a11, 1, 1);
  store_tile(a02, 0, 2); store_tile(a12, 1, 2);
}

// ---------------------------------------------------------------------------
extern "C" void kernel_launch(void* const* d_in, const int* in_sizes, int n_in,
                              void* d_out, int out_size, void* d_ws, size_t ws_size,
                              hipStream_t stream) {
  const float* hs     = (const float*)d_in[0];
  const float* W_attn = (const float*)d_in[1];
  const float* b_attn = (const float*)d_in[2];
  const float* P_q    = (const float*)d_in[3];
  const float* P_k    = (const float*)d_in[4];
  const float* P_v    = (const float*)d_in[5];
  const float* W_back = (const float*)d_in[6];
  const float* W_proj = (const float*)d_in[7];
  const float* b_proj = (const float*)d_in[8];

  char* ws = (char*)d_ws;                             // ~27.6 MB total
  half_t* WqkvT = (half_t*)ws;  ws += (size_t)N3D * ND * sizeof(half_t);
  half_t* WoutT = (half_t*)ws;  ws += (size_t)ND * NDC * sizeof(half_t);
  float*  b_eff = (float*)ws;   ws += (size_t)N3D * sizeof(float);
  ws = (char*)(((uintptr_t)ws + 255) & ~(uintptr_t)255);
  half_t* Qh  = (half_t*)ws;    ws += (size_t)NB * NH * NS * NHD * sizeof(half_t);
  half_t* Kh  = (half_t*)ws;    ws += (size_t)NB * NH * NS * NHD * sizeof(half_t);
  half_t* Vt  = (half_t*)ws;    ws += (size_t)NB * NH * NHD * NS * sizeof(half_t);
  half_t* Ctx = (half_t*)ws;    ws += (size_t)NROWS * NDC * sizeof(half_t);

  {
    int total = N3D * ND + ND * NDC + N3D;
    fold_kernel<<<(total + 255) / 256, 256, 0, stream>>>(
        W_attn, b_attn, P_q, P_k, P_v, W_back, W_proj, WqkvT, WoutT, b_eff);
  }
  qkv_gemm_kernel<<<dim3(NROWS / 32, 6), 128, 0, stream>>>(
      hs, WqkvT, b_eff, Qh, Kh, Vt);
  attn_kernel<<<dim3(NB * NH, NS / 16), 32, 0, stream>>>(Qh, Kh, Vt, Ctx);
  out_gemm_kernel<<<dim3(NROWS / 32, 4), 128, 0, stream>>>(
      Ctx, WoutT, b_proj, (float*)d_out);
}
